// GraphAttentionLayer_25469156065427
// MI455X (gfx1250) — matrix operand order repaired and checked by reading, not compiled, not run on hardware
//
#include <hip/hip_runtime.h>
#include <hip/hip_bf16.h>

typedef __attribute__((ext_vector_type(16))) _Float16 v16h;
typedef __attribute__((ext_vector_type(8)))  float    v8f;
typedef __attribute__((ext_vector_type(4)))  float    v4f;
typedef __attribute__((ext_vector_type(4)))  int      v4i;

#define GAT_N    8192
#define GAT_INF  256
#define GAT_OUTF 64
#define GAT_ALPHA 0.2f

// ---------------------------------------------------------------------------
// Kernel 1: Wh = h @ W via WMMA f16 (f32 accum). Emits WhT (f16, [64][8192])
// and e1 = Wh@a1, e2 = Wh@a2 (f32). One wave handles a 16-row tile.
// ---------------------------------------------------------------------------
__global__ void k_wh(const float* __restrict__ h, const float* __restrict__ W,
                     const float* __restrict__ a1, const float* __restrict__ a2,
                     _Float16* __restrict__ WhT, float* __restrict__ e1,
                     float* __restrict__ e2)
{
    // W staged transposed in LDS as f16: Wl[n][k], padded row so 32B-aligned.
    __shared__ __align__(32) _Float16 Wl[GAT_OUTF][GAT_INF + 16];
    for (int idx = threadIdx.x; idx < GAT_INF * GAT_OUTF; idx += blockDim.x) {
        int k = idx >> 6, n = idx & 63;
        Wl[n][k] = (_Float16)W[idx];
    }
    __syncthreads();

    const int wave  = threadIdx.x >> 5;
    const int lane  = threadIdx.x & 31;
    const int row16 = lane & 15;
    const int half  = lane >> 4;
    const int i0    = (blockIdx.x * 4 + wave) * 16;
    const int row   = i0 + row16;

    v8f acc[4] = {};
    for (int k0 = 0; k0 < GAT_INF; k0 += 32) {
        // A tile: h rows, f32 -> f16. Lane layout: elems 0..15 = K 16h..16h+15.
        const float* hp = h + (size_t)row * GAT_INF + k0 + half * 16;
        v4f h0 = *(const v4f*)(hp +  0);
        v4f h1 = *(const v4f*)(hp +  4);
        v4f h2 = *(const v4f*)(hp +  8);
        v4f h3 = *(const v4f*)(hp + 12);
        v16h a;
#pragma unroll
        for (int t = 0; t < 4; ++t) {
            a[t]      = (_Float16)h0[t];
            a[4 + t]  = (_Float16)h1[t];
            a[8 + t]  = (_Float16)h2[t];
            a[12 + t] = (_Float16)h3[t];
        }
#pragma unroll
        for (int nt = 0; nt < 4; ++nt) {
            // B elem k = W[k0 + k + 16*half][nt*16 + row16]
            v16h b = *(const v16h*)&Wl[nt * 16 + row16][k0 + half * 16];
            acc[nt] = __builtin_amdgcn_wmma_f32_16x16x32_f16(
                false, a, false, b, (short)0, acc[nt], false, false);
        }
    }

    // D layout: VGPR v -> row (v + 8*half), col (nt*16 + row16).
    float p1[8], p2[8];
#pragma unroll
    for (int v = 0; v < 8; ++v) { p1[v] = 0.f; p2[v] = 0.f; }
#pragma unroll
    for (int nt = 0; nt < 4; ++nt) {
        int col = nt * 16 + row16;
        float A1 = a1[col], A2 = a2[col];
#pragma unroll
        for (int v = 0; v < 8; ++v) {
            int r = i0 + half * 8 + v;
            float val = acc[nt][v];
            WhT[(size_t)col * GAT_N + r] = (_Float16)val;
            p1[v] += val * A1;
            p2[v] += val * A2;
        }
    }
    // Reduce over the 16 columns held by each lane group (xor stays in-group).
#pragma unroll
    for (int o = 1; o < 16; o <<= 1) {
#pragma unroll
        for (int v = 0; v < 8; ++v) {
            p1[v] += __shfl_xor(p1[v], o, 32);
            p2[v] += __shfl_xor(p2[v], o, 32);
        }
    }
    if (row16 == 0) {
#pragma unroll
        for (int v = 0; v < 8; ++v) {
            e1[i0 + half * 8 + v] = p1[v];
            e2[i0 + half * 8 + v] = p2[v];
        }
    }
}

// ---------------------------------------------------------------------------
// Kernel 2: global max of e2 (single block)
// ---------------------------------------------------------------------------
__global__ void k_e2max(const float* __restrict__ e2, float* __restrict__ e2m)
{
    __shared__ float s[256];
    float m = -3.0e38f;
    for (int i = threadIdx.x; i < GAT_N; i += 256) m = fmaxf(m, e2[i]);
    s[threadIdx.x] = m;
    __syncthreads();
    for (int o = 128; o > 0; o >>= 1) {
        if ((int)threadIdx.x < o) s[threadIdx.x] = fmaxf(s[threadIdx.x], s[threadIdx.x + o]);
        __syncthreads();
    }
    if (threadIdx.x == 0) *e2m = s[0];
}

// ---------------------------------------------------------------------------
// Kernel 3: fused masked-softmax attention @ Wh, + ELU.
// Block = 4 waves, one 16-row i-tile per block; each wave owns a 2048-wide
// j range. Weights computed on the fly (rank-1 logits + adj mask), exp'd with
// a safe per-row upper bound m = leaky(e1[i] + max(e2)) so w in (0,1] (f16-ok).
// 4x v_wmma_f32_16x16x32_f16 per 32-j step. Partials combined through LDS.
// ---------------------------------------------------------------------------
__global__ void k_attn(const int* __restrict__ adj,
                       const float* __restrict__ e1,
                       const float* __restrict__ e2,
                       const float* __restrict__ e2m,
                       const _Float16* __restrict__ WhT,
                       float* __restrict__ out)
{
    __shared__ float zsh[4][16];
    __shared__ float accsh[4][32][32];   // [wave][lane][nt*8+v]

    const int wave  = threadIdx.x >> 5;
    const int lane  = threadIdx.x & 31;
    const int row16 = lane & 15;
    const int half  = lane >> 4;
    const int i0    = blockIdx.x * 16;
    const int row   = i0 + row16;

    const float gmax = *e2m;
    const float e1r  = e1[row];
    float mb = e1r + gmax;
    mb = (mb > 0.f) ? mb : GAT_ALPHA * mb;   // row upper bound of logits

    const size_t adjrow = (size_t)row * GAT_N;
    const int jbase = wave * 2048;

    v8f acc[4] = {};
    float z = 0.f;

    for (int jt = 0; jt < 64; ++jt) {
        const int j0 = jbase + jt * 32;
        const int* ap  = adj + adjrow + j0 + half * 16;
        const float* ep = e2 + j0 + half * 16;

        v16h aw;
#pragma unroll
        for (int c = 0; c < 2; ++c) {        // c=0: K = 16h+0..7 ; c=1: +8
            const int off = c * 8;
            v4i m0 = *(const v4i*)(ap + off);
            v4i m1 = *(const v4i*)(ap + off + 4);
            v4f f0 = *(const v4f*)(ep + off);
            v4f f1 = *(const v4f*)(ep + off + 4);
#pragma unroll
            for (int t = 0; t < 4; ++t) {
                float s0 = e1r + f0[t];
                s0 = (s0 > 0.f) ? s0 : GAT_ALPHA * s0;
                float w0 = (m0[t] > 0) ? __expf(s0 - mb) : 0.f;
                float s1 = e1r + f1[t];
                s1 = (s1 > 0.f) ? s1 : GAT_ALPHA * s1;
                float w1 = (m1[t] > 0) ? __expf(s1 - mb) : 0.f;
                z += w0 + w1;
                aw[c * 8 + t]     = (_Float16)w0;
                aw[c * 8 + 4 + t] = (_Float16)w1;
            }
        }
#pragma unroll
        for (int nt = 0; nt < 4; ++nt) {
            // B elem k = Wh[j0 + k + 16*half][nt*16 + row16] = WhT row read
            v16h b = *(const v16h*)(WhT + (size_t)(nt * 16 + row16) * GAT_N
                                        + j0 + half * 16);
            acc[nt] = __builtin_amdgcn_wmma_f32_16x16x32_f16(
                false, aw, false, b, (short)0, acc[nt], false, false);
        }
    }

    // Row-sum of z within the wave (lanes l and l^16 hold the same row).
    z += __shfl_xor(z, 16, 32);
    if (half == 0) zsh[wave][row16] = z;
#pragma unroll
    for (int nt = 0; nt < 4; ++nt)
#pragma unroll
        for (int v = 0; v < 8; ++v) accsh[wave][lane][nt * 8 + v] = acc[nt][v];
    __syncthreads();

    if (wave == 0) {
        float Zrow[8];
#pragma unroll
        for (int v = 0; v < 8; ++v) {
            int ri = v + 8 * half;
            Zrow[v] = zsh[0][ri] + zsh[1][ri] + zsh[2][ri] + zsh[3][ri];
        }
#pragma unroll
        for (int nt = 0; nt < 4; ++nt) {
            int col = nt * 16 + row16;
#pragma unroll
            for (int v = 0; v < 8; ++v) {
                float t = (accsh[0][lane][nt * 8 + v] + accsh[1][lane][nt * 8 + v] +
                           accsh[2][lane][nt * 8 + v] + accsh[3][lane][nt * 8 + v]) / Zrow[v];
                float o = (t > 0.f) ? t : (__expf(t) - 1.f);   // ELU
                out[(size_t)(i0 + 8 * half + v) * GAT_OUTF + col] = o;
            }
        }
    }
}

// ---------------------------------------------------------------------------
extern "C" void kernel_launch(void* const* d_in, const int* in_sizes, int n_in,
                              void* d_out, int out_size, void* d_ws, size_t ws_size,
                              hipStream_t stream)
{
    const float* h   = (const float*)d_in[0];
    const int*   adj = (const int*)  d_in[1];
    const float* W   = (const float*)d_in[2];
    const float* a1  = (const float*)d_in[3];
    const float* a2  = (const float*)d_in[4];
    float* out = (float*)d_out;

    char* ws = (char*)d_ws;
    _Float16* WhT = (_Float16*)ws;                         // 64*8192*2 = 1 MB
    float* e1  = (float*)(ws + (size_t)GAT_OUTF * GAT_N * sizeof(_Float16));
    float* e2  = e1 + GAT_N;
    float* e2m = e2 + GAT_N;

    k_wh  <<<GAT_N / 64, 128, 0, stream>>>(h, W, a1, a2, WhT, e1, e2);
    k_e2max<<<1, 256, 0, stream>>>(e2, e2m);
    k_attn<<<GAT_N / 16, 128, 0, stream>>>(adj, e1, e2, e2m, WhT, out);
}
